// SwinTransformerBlock_75969381532249
// MI455X (gfx1250) — matrix operand order repaired and checked by reading, not compiled, not run on hardware
//
#include <hip/hip_runtime.h>
#include <math.h>

// ---------------------------------------------------------------------------
// Swin Transformer block for MI455X (gfx1250), wave32, bf16 WMMA pipeline.
//   ln1 -> QKV GEMM -> window attention (1 wave / window-head) -> proj GEMM
//   -> residual+ln2 -> MLP GEMM1 (GELU) -> MLP GEMM2 -> residual + transpose
// All GEMMs use v_wmma_f32_16x16x32_bf16 with fp32 accumulation.
// ---------------------------------------------------------------------------

typedef __bf16 v16bf __attribute__((ext_vector_type(16)));
typedef float v8f   __attribute__((ext_vector_type(8)));

#define Bsz 4
#define Cch 96
#define Hdim 224
#define Wdim 224
#define WS 7
#define NH 3
#define HD 32
#define NT 49
#define HID 384
#define HWp (Hdim * Wdim)              // 50176
#define NWIN (Bsz * 32 * 32)           // 4096 windows
#define TTOK (NWIN * NT)               // 200704 tokens
#define QK_SCALE 0.17677669529663687f  // 32^-0.5

union FragU { v16bf v; unsigned int u[8]; };

__device__ inline unsigned short f2bf(float f) {
    unsigned int x = __float_as_uint(f);
    unsigned int r = x + 0x7fffu + ((x >> 16) & 1u);   // round-nearest-even
    return (unsigned short)(r >> 16);
}

// A fragment 16x32 bf16 (ISA 7.12.2): lanes 0-15 -> M=lane, K {0..7,16..23};
// lanes 16-31 -> M=lane-16, K {8..15,24..31}. 2 bf16 per VGPR.
__device__ inline v16bf load_a_frag(const unsigned short* lds, int m0,
                                    int pitchU, int kuOff) {
    const int lane = threadIdx.x & 31;
    const int m = m0 + (lane & 15);
    const int kb = (lane < 16) ? 0 : 4;                // uint (k-pair) base
    const unsigned int* p = (const unsigned int*)lds;
    FragU f;
#pragma unroll
    for (int vg = 0; vg < 8; ++vg) {
        int ku = kuOff + kb + (vg & 3) + ((vg >> 2) << 3);
        f.u[vg] = p[m * pitchU + ku];
    }
    return f.v;
}

// B fragment 32x16 bf16: lanes 0-15 -> N=lane, K {0..7,16..23}; lanes 16-31 ->
// N=lane-16, K {8..15,24..31} (same K order as A). LDS holds B
// transposed: row n, K contiguous -> k-pairs contiguous.
__device__ inline v16bf load_b_frag(const unsigned short* lds, int n0,
                                    int pitchU, int kuOff) {
    const int lane = threadIdx.x & 31;
    const int n = n0 + (lane & 15);
    const int kb = (lane < 16) ? 0 : 4;
    const unsigned int* p = (const unsigned int*)lds;
    FragU f;
#pragma unroll
    for (int vg = 0; vg < 8; ++vg)
        f.u[vg] = p[n * pitchU + kb + (vg & 3) + ((vg >> 2) << 3) + kuOff];
    return f.v;
}

__device__ inline v8f wmma_bf16(v16bf a, v16bf b, v8f c) {
    return __builtin_amdgcn_wmma_f32_16x16x32_bf16(
        false, a, false, b, (short)0, c, false, false);
}

// ---------------------------------------------------------------------------
__global__ void cvt_bf16_kernel(const float* __restrict__ src,
                                unsigned short* __restrict__ dst, int n) {
    int i = blockIdx.x * 256 + threadIdx.x;
    if (i < n) dst[i] = f2bf(src[i]);
}

// ---------------------------------------------------------------------------
// LayerNorm1: NCHW f32 -> bf16 [T,96] in window-token order.
__global__ __launch_bounds__(224) void ln1_kernel(
    const float* __restrict__ x, const float* __restrict__ g,
    const float* __restrict__ bta, unsigned short* __restrict__ Xln) {
    const int w = threadIdx.x;
    const int b = blockIdx.x / Hdim, hh = blockIdx.x % Hdim;
    const long base0 = (long)b * Cch * HWp + hh * Wdim + w;
    float sum = 0.f, sq = 0.f;
#pragma unroll 4
    for (int c = 0; c < Cch; ++c) {
        float v = x[base0 + (long)c * HWp];
        sum += v; sq += v * v;
    }
    float mean = sum * (1.f / Cch);
    float var = sq * (1.f / Cch) - mean * mean;
    float rstd = rsqrtf(var + 1e-5f);
    const int wh = hh / WS, ii = hh % WS, ww = w / WS, jj = w % WS;
    const long row = ((long)(b * 32 + wh) * 32 + ww) * NT + ii * WS + jj;
#pragma unroll 4
    for (int c = 0; c < Cch; ++c) {
        float v = x[base0 + (long)c * HWp];
        Xln[row * Cch + c] = f2bf((v - mean) * rstd * g[c] + bta[c]);
    }
}

// ---------------------------------------------------------------------------
// Generic bf16 GEMM: C[T,Ncols] = A[T,K] * Bw[K,Ncols] (+epilogue).
// Block tile 128(M) x 32(N), 8 waves of 16x32, K step 32 (one WMMA pair).
// EPI 0: +bias, split-scatter to Q(*scale)/K/V bf16 [nW,NH,49,32]
// EPI 1: +bias, f32 out                  EPI 2: +bias, exact GELU, bf16 out
template <int EPI>
__global__ __launch_bounds__(256) void gemm_kernel(
    const unsigned short* __restrict__ A, const unsigned short* __restrict__ Bw,
    int K, int Ncols, const float* __restrict__ bias,
    unsigned short* __restrict__ o_q, unsigned short* __restrict__ o_k,
    unsigned short* __restrict__ o_v, float* __restrict__ o_f,
    unsigned short* __restrict__ o_h) {
    __shared__ __align__(16) unsigned short As[128 * 40];  // pitch 40 us = 20 u
    __shared__ __align__(16) unsigned short Bs[32 * 40];   // transposed [n][k]
    const int tid = threadIdx.x;
    const int wv = tid >> 5, lane = tid & 31;
    const long mBase = (long)blockIdx.x * 128;
    const int nBase = blockIdx.y * 32;

    v8f acc[2] = {};
    for (int k0 = 0; k0 < K; k0 += 32) {
        __syncthreads();
#pragma unroll
        for (int it = 0; it < 2; ++it) {          // A tile: 128x32, uint4 loads
            int idx = it * 256 + tid;             // 0..511
            int r = idx >> 2, q = idx & 3;
            const uint4* src =
                (const uint4*)(A + (mBase + r) * K + k0 + q * 8);
            *(uint4*)(&As[r * 40 + q * 8]) = *src;
        }
#pragma unroll
        for (int it = 0; it < 4; ++it) {          // B tile: 32x32 -> transposed
            int idx = it * 256 + tid;             // 0..1023
            int n = idx & 31, k = idx >> 5;
            Bs[n * 40 + k] = Bw[(long)(k0 + k) * Ncols + nBase + n];
        }
        __syncthreads();
        v16bf af = load_a_frag(As, wv * 16, 20, 0);
        v16bf b0 = load_b_frag(Bs, 0, 20, 0);
        v16bf b1 = load_b_frag(Bs, 16, 20, 0);
        acc[0] = wmma_bf16(af, b0, acc[0]);
        acc[1] = wmma_bf16(af, b1, acc[1]);
    }

    const int half8 = (lane >> 4) << 3, nl = lane & 15;
    if (EPI == 0) {
        // block-uniform: which of q/k/v and which head this 32-col slab is
        const int s = nBase / 96;                 // 0=q 1=k 2=v
        const int h = (nBase % 96) >> 5;          // head
        unsigned short* dst = (s == 0) ? o_q : (s == 1) ? o_k : o_v;
        const float scl = (s == 0) ? QK_SCALE : 1.0f;
#pragma unroll
        for (int j = 0; j < 2; ++j) {
#pragma unroll
            for (int r = 0; r < 8; ++r) {
                long row = mBase + wv * 16 + r + half8;
                int col = nBase + j * 16 + nl;
                int d = j * 16 + nl;
                float val = (acc[j][r] + bias[col]) * scl;
                long wdw = row / NT; int i = (int)(row % NT);
                dst[((wdw * NH + h) * NT + i) * HD + d] = f2bf(val);
            }
        }
    } else {
#pragma unroll
        for (int j = 0; j < 2; ++j) {
#pragma unroll
            for (int r = 0; r < 8; ++r) {
                long row = mBase + wv * 16 + r + half8;
                int col = nBase + j * 16 + nl;
                float val = acc[j][r] + bias[col];
                if (EPI == 1) {
                    o_f[row * Ncols + col] = val;
                } else {
                    float gl =
                        0.5f * val * (1.0f + erff(val * 0.70710678118f));
                    o_h[row * Ncols + col] = f2bf(gl);
                }
            }
        }
    }
}

// ---------------------------------------------------------------------------
// Window attention: one wave per (window, head). S=Q*K^T (+rel bias), softmax
// via cross-lane shuffles inside 16-lane halves, then P*V. All WMMA bf16.
__global__ __launch_bounds__(32) void attn_kernel(
    const unsigned short* __restrict__ Qg, const unsigned short* __restrict__ Kg,
    const unsigned short* __restrict__ Vg, const float* __restrict__ rel_bias,
    const int* __restrict__ rel_index, unsigned short* __restrict__ AO) {
    __shared__ __align__(16) unsigned short qs[64 * 34];   // [m][d] pitch 17u
    __shared__ __align__(16) unsigned short ks[64 * 34];   // [j][d] pitch 17u
    __shared__ __align__(16) unsigned short vst[32 * 66];  // [d][j] pitch 33u
    __shared__ __align__(16) unsigned short ps[16 * 66];   // [i][j] pitch 33u
    __shared__ __align__(16) unsigned short os[16 * 34];   // out staging
    const int lane = threadIdx.x;
    const int wdw = blockIdx.x / NH, h = blockIdx.x % NH;
    const long base = ((long)wdw * NH + h) * NT * HD;

    for (int i = 0; i < 64; ++i) {                 // stage q/k/v, zero-pad
        unsigned short qv = 0, kv = 0, vv = 0;
        if (i < NT) {                              // uniform branch
            qv = Qg[base + i * HD + lane];
            kv = Kg[base + i * HD + lane];
            vv = Vg[base + i * HD + lane];
        }
        qs[i * 34 + lane] = qv;
        ks[i * 34 + lane] = kv;
        vst[lane * 66 + i] = vv;
    }
    __syncthreads();

    const int half = lane >> 4, jl = lane & 15;
    for (int mt = 0; mt < 4; ++mt) {
        v8f sa[4] = {};
        v16bf aq = load_a_frag(qs, mt * 16, 17, 0);
#pragma unroll
        for (int nt = 0; nt < 4; ++nt) {
            v16bf bk = load_b_frag(ks, nt * 16, 17, 0);
            sa[nt] = wmma_bf16(aq, bk, sa[nt]);
        }
        // softmax over j (49 cols): lane owns rows r+8*half, cols jl+16*nt.
        // Branch-free bias gather: clamp indices, select -inf for pad cols.
#pragma unroll
        for (int r = 0; r < 8; ++r) {
            int i = mt * 16 + r + half * 8;
            int iC = (i < NT) ? i : (NT - 1);
            float sv[4];
#pragma unroll
            for (int nt = 0; nt < 4; ++nt) {
                int j = nt * 16 + jl;
                int jC = (j < NT) ? j : (NT - 1);
                float bv = rel_bias[rel_index[iC * NT + jC] * NH + h];
                float v = sa[nt][r] + bv;
                sv[nt] = (j < NT) ? v : -1e30f;
            }
            float mx = fmaxf(fmaxf(sv[0], sv[1]), fmaxf(sv[2], sv[3]));
#pragma unroll
            for (int off = 8; off >= 1; off >>= 1)
                mx = fmaxf(mx, __shfl_xor(mx, off, 32));
            float sum = 0.f;
#pragma unroll
            for (int nt = 0; nt < 4; ++nt) {
                sv[nt] = __expf(sv[nt] - mx);
                sum += sv[nt];
            }
#pragma unroll
            for (int off = 8; off >= 1; off >>= 1)
                sum += __shfl_xor(sum, off, 32);
            float inv = 1.0f / sum;
#pragma unroll
            for (int nt = 0; nt < 4; ++nt)
                ps[(r + half * 8) * 66 + nt * 16 + jl] = f2bf(sv[nt] * inv);
        }
        __syncthreads();  // P visible before fragment reads
        // P[16x64] * V[64x32]
        v8f oa[2] = {};
#pragma unroll
        for (int kk = 0; kk < 2; ++kk) {
            v16bf ap = load_a_frag(ps, 0, 33, kk * 16);
#pragma unroll
            for (int jf = 0; jf < 2; ++jf) {
                v16bf bv = load_b_frag(vst, jf * 16, 33, kk * 16);
                oa[jf] = wmma_bf16(ap, bv, oa[jf]);
            }
        }
        // stage output tile to LDS, then row-wise coalesced stores with a
        // wave-uniform validity guard (no per-lane exec divergence).
#pragma unroll
        for (int jf = 0; jf < 2; ++jf)
#pragma unroll
            for (int r = 0; r < 8; ++r)
                os[(r + half * 8) * 34 + jf * 16 + jl] = f2bf(oa[jf][r]);
        __syncthreads();
#pragma unroll
        for (int rr = 0; rr < 16; ++rr) {
            int i = mt * 16 + rr;
            if (i < NT)  // uniform across the wave
                AO[((long)wdw * NT + i) * Cch + h * HD + lane] =
                    os[rr * 34 + lane];
        }
        __syncthreads();  // before ps/os are overwritten next iteration
    }
}

// ---------------------------------------------------------------------------
// x1 = x + proj_out (window order) ; ln2(x1) -> bf16 [T,96] raster order.
__global__ __launch_bounds__(224) void res_ln2_kernel(
    const float* __restrict__ x, const float* __restrict__ Pout,
    const float* __restrict__ g, const float* __restrict__ bta,
    float* __restrict__ x1, unsigned short* __restrict__ Xln2) {
    const int w = threadIdx.x;
    const int b = blockIdx.x / Hdim, hh = blockIdx.x % Hdim;
    const long base0 = (long)b * Cch * HWp + hh * Wdim + w;
    const int wh = hh / WS, ii = hh % WS, ww = w / WS, jj = w % WS;
    const long row = ((long)(b * 32 + wh) * 32 + ww) * NT + ii * WS + jj;
    const long t = ((long)b * Hdim + hh) * Wdim + w;
    float sum = 0.f, sq = 0.f;
#pragma unroll 4
    for (int c = 0; c < Cch; ++c) {
        float v = x[base0 + (long)c * HWp] + Pout[row * Cch + c];
        x1[t * Cch + c] = v;
        sum += v; sq += v * v;
    }
    float mean = sum * (1.f / Cch);
    float var = sq * (1.f / Cch) - mean * mean;
    float rstd = rsqrtf(var + 1e-5f);
#pragma unroll 4
    for (int c = 0; c < Cch; ++c) {
        float v = x1[t * Cch + c];
        Xln2[t * Cch + c] = f2bf((v - mean) * rstd * g[c] + bta[c]);
    }
}

// ---------------------------------------------------------------------------
// out(NCHW) = x1 + mlp_out  (both [T,96] raster order)
__global__ __launch_bounds__(224) void final_kernel(
    const float* __restrict__ x1, const float* __restrict__ Mout,
    float* __restrict__ out) {
    const int w = threadIdx.x;
    const int b = blockIdx.x / Hdim, hh = blockIdx.x % Hdim;
    const long t = ((long)b * Hdim + hh) * Wdim + w;
#pragma unroll 4
    for (int c = 0; c < Cch; ++c)
        out[((long)(b * Cch + c)) * HWp + hh * Wdim + w] =
            x1[t * Cch + c] + Mout[t * Cch + c];
}

// ---------------------------------------------------------------------------
extern "C" void kernel_launch(void* const* d_in, const int* in_sizes, int n_in,
                              void* d_out, int out_size, void* d_ws,
                              size_t ws_size, hipStream_t stream) {
    (void)in_sizes; (void)n_in; (void)out_size; (void)ws_size;
    const float* x       = (const float*)d_in[0];
    const float* n1g     = (const float*)d_in[1];
    const float* n1b     = (const float*)d_in[2];
    const float* qkv_w   = (const float*)d_in[3];
    const float* qkv_b   = (const float*)d_in[4];
    const float* proj_w  = (const float*)d_in[5];
    const float* proj_b  = (const float*)d_in[6];
    const float* rel_b   = (const float*)d_in[7];
    const float* n2g     = (const float*)d_in[8];
    const float* n2b     = (const float*)d_in[9];
    const float* mlp_w1  = (const float*)d_in[10];
    const float* mlp_b1  = (const float*)d_in[11];
    const float* mlp_w2  = (const float*)d_in[12];
    const float* mlp_b2  = (const float*)d_in[13];
    const int*   rel_idx = (const int*)d_in[14];

    char* ws = (char*)d_ws;
    // weights (bf16)
    unsigned short* Wq = (unsigned short*)(ws + 0);        //  55296 B
    unsigned short* Wp = (unsigned short*)(ws + 65536);    //  18432 B
    unsigned short* W1 = (unsigned short*)(ws + 98304);    //  73728 B
    unsigned short* W2 = (unsigned short*)(ws + 196608);   //  73728 B
    // region A (77 MB): Xln -> attn_out -> Xln2 -> Mout (sequential reuse)
    char* rA = ws + (size_t)512 * 1024;
    unsigned short* Xln  = (unsigned short*)rA;
    unsigned short* AOut = (unsigned short*)rA;
    unsigned short* Xln2 = (unsigned short*)rA;
    float*          Mout = (float*)rA;
    // region B (154 MB): QKV -> Pout -> H (sequential reuse)
    char* rB = ws + (size_t)80 * 1024 * 1024;
    unsigned short* Qb = (unsigned short*)rB;
    unsigned short* Kb = (unsigned short*)(rB + (size_t)TTOK * Cch * 2);
    unsigned short* Vb = (unsigned short*)(rB + (size_t)TTOK * Cch * 4);
    float*          Pout = (float*)rB;
    unsigned short* Hbuf = (unsigned short*)rB;
    // region C (77 MB): x1
    float* X1 = (float*)(ws + (size_t)240 * 1024 * 1024);

    // 1) convert weights to bf16
    cvt_bf16_kernel<<<(27648 + 255) / 256, 256, 0, stream>>>(qkv_w, Wq, 27648);
    cvt_bf16_kernel<<<(9216 + 255) / 256, 256, 0, stream>>>(proj_w, Wp, 9216);
    cvt_bf16_kernel<<<(36864 + 255) / 256, 256, 0, stream>>>(mlp_w1, W1, 36864);
    cvt_bf16_kernel<<<(36864 + 255) / 256, 256, 0, stream>>>(mlp_w2, W2, 36864);

    const int MB = TTOK / 128;  // 1568
    // 2) LN1
    ln1_kernel<<<Bsz * Hdim, 224, 0, stream>>>(x, n1g, n1b, Xln);
    // 3) QKV GEMM [T,96]x[96,288]
    gemm_kernel<0><<<dim3(MB, 9), 256, 0, stream>>>(
        Xln, Wq, 96, 288, qkv_b, Qb, Kb, Vb, nullptr, nullptr);
    // 4) window attention
    attn_kernel<<<NWIN * NH, 32, 0, stream>>>(Qb, Kb, Vb, rel_b, rel_idx, AOut);
    // 5) proj GEMM [T,96]x[96,96]
    gemm_kernel<1><<<dim3(MB, 3), 256, 0, stream>>>(
        AOut, Wp, 96, 96, proj_b, nullptr, nullptr, nullptr, Pout, nullptr);
    // 6) residual + LN2
    res_ln2_kernel<<<Bsz * Hdim, 224, 0, stream>>>(x, Pout, n2g, n2b, X1, Xln2);
    // 7) MLP GEMM1 + GELU [T,96]x[96,384]
    gemm_kernel<2><<<dim3(MB, 12), 256, 0, stream>>>(
        Xln2, W1, 96, 384, mlp_b1, nullptr, nullptr, nullptr, nullptr, Hbuf);
    // 8) MLP GEMM2 [T,384]x[384,96]
    gemm_kernel<1><<<dim3(MB, 3), 256, 0, stream>>>(
        Hbuf, W2, 384, 96, mlp_b2, nullptr, nullptr, nullptr, Mout, nullptr);
    // 9) final residual + NHWC->NCHW
    final_kernel<<<Bsz * Hdim, 224, 0, stream>>>(X1, Mout, (float*)d_out);
}
